// Segmentation_58162447123092
// MI455X (gfx1250) — matrix operand order repaired and checked by reading, not compiled, not run on hardware
//
#include <hip/hip_runtime.h>

typedef __attribute__((ext_vector_type(16))) __bf16 v16bf;
typedef __attribute__((ext_vector_type(8)))  float  v8f;
typedef __attribute__((ext_vector_type(2)))  float  v2f;

#define NB   4096
#define NN   64
#define ND   64
#define NF   4096
#define KP   2080   // 64*65/2, multiple of 32
#define KSTEPS (KP / 32)

// ---------------------------------------------------------------------------
// CDNA5 async copy: global -> LDS, 16 bytes per lane, tracked by ASYNCcnt.
// vdst = LDS byte address (wave-relative), vaddr = 64-bit global address.
// ---------------------------------------------------------------------------
__device__ __forceinline__ void async_copy_b128(unsigned lds_addr, const void* gaddr) {
  asm volatile("global_load_async_to_lds_b128 %0, %1, off"
               :: "v"(lds_addr), "v"(gaddr) : "memory");
}
__device__ __forceinline__ void wait_async_le4() {
  asm volatile("s_wait_asynccnt 4" ::: "memory");
}
__device__ __forceinline__ void wait_async_0() {
  asm volatile("s_wait_asynccnt 0" ::: "memory");
}

// ---------------------------------------------------------------------------
// Kernel 1: per-batch  T = E @ kp_W ; S = T @ E^T ; extract triu/tril as bf16;
//           copy E to embs_flatten.  One block (4 waves) per batch.
// Uses V_WMMA_F32_16X16X4_F32 for the small f32 GEMM chains.
// ---------------------------------------------------------------------------
__global__ __launch_bounds__(128) void kp_extract_kernel(
    const float* __restrict__ Eg, const float* __restrict__ Wg,
    float* __restrict__ flat, __bf16* __restrict__ triu, __bf16* __restrict__ tril)
{
  __shared__ float Es[64][66];   // padded rows -> bank spread
  __shared__ float Ws[64][66];
  __shared__ float Ts[64][66];

  const int b    = blockIdx.x;
  const int t    = threadIdx.x;     // 0..127
  const int wid  = t >> 5;          // wave 0..3
  const int lane = t & 31;
  const int rl   = lane & 15;       // lane row/col within 16
  const int q    = lane >> 4;       // half-wave select

  // ---- load E into LDS + stream copy to embs_flatten; load kp_W into LDS ----
  const float4* Eb = (const float4*)(Eg + (size_t)b * (NN * ND));
  float4*       Fb = (float4*)(flat + (size_t)b * (NN * ND));
#pragma unroll
  for (int i = 0; i < 8; ++i) {
    const int idx = t + 128 * i;            // 1024 float4 = 64x64 floats
    const float4 v = Eb[idx];
    Fb[idx] = v;
    const int r = idx >> 4, c = (idx & 15) << 2;
    Es[r][c + 0] = v.x; Es[r][c + 1] = v.y; Es[r][c + 2] = v.z; Es[r][c + 3] = v.w;
  }
#pragma unroll
  for (int i = 0; i < 8; ++i) {
    const int idx = t + 128 * i;
    const float4 v = ((const float4*)Wg)[idx];
    const int r = idx >> 4, c = (idx & 15) << 2;
    Ws[r][c + 0] = v.x; Ws[r][c + 1] = v.y; Ws[r][c + 2] = v.z; Ws[r][c + 3] = v.w;
  }
  __syncthreads();

  const v8f vz = {0.f, 0.f, 0.f, 0.f, 0.f, 0.f, 0.f, 0.f};
  const int mrow = 16 * wid + rl;   // this wave's 16-row strip

  // ---- phase 2: T(strip) = E(strip) @ W  (K = 64, 16 steps of K=4) ----
  v8f tacc[4] = {vz, vz, vz, vz};
  for (int ks = 0; ks < 16; ++ks) {
    const int k0 = ks * 4 + q * 2;
    v2f a; a.x = Es[mrow][k0]; a.y = Es[mrow][k0 + 1];
#pragma unroll
    for (int nt = 0; nt < 4; ++nt) {
      v2f bb; bb.x = Ws[k0][nt * 16 + rl]; bb.y = Ws[k0 + 1][nt * 16 + rl];
      tacc[nt] = __builtin_amdgcn_wmma_f32_16x16x4_f32(
          false, a, false, bb, (short)0, tacc[nt], false, false);
    }
  }
  // spill T strip to LDS (wave-local round trip; each wave reads only its rows)
#pragma unroll
  for (int nt = 0; nt < 4; ++nt)
#pragma unroll
    for (int r = 0; r < 8; ++r)
      Ts[16 * wid + r + 8 * q][nt * 16 + rl] = tacc[nt][r];

  // ---- phase 3: S(strip) = T(strip) @ E^T ----
  v8f sacc[4] = {vz, vz, vz, vz};
  for (int ks = 0; ks < 16; ++ks) {
    const int k0 = ks * 4 + q * 2;
    v2f a; a.x = Ts[mrow][k0]; a.y = Ts[mrow][k0 + 1];
#pragma unroll
    for (int nt = 0; nt < 4; ++nt) {
      const int m = nt * 16 + rl;
      v2f bb; bb.x = Es[m][k0]; bb.y = Es[m][k0 + 1];   // B[k][m] = E[m][k]
      sacc[nt] = __builtin_amdgcn_wmma_f32_16x16x4_f32(
          false, a, false, bb, (short)0, sacc[nt], false, false);
    }
  }

  // ---- phase 4: scatter S into triu/tril (bf16) ----
  const size_t base = (size_t)b * KP;
#pragma unroll
  for (int nt = 0; nt < 4; ++nt) {
    const int j = nt * 16 + rl;
#pragma unroll
    for (int r = 0; r < 8; ++r) {
      const int i = 16 * wid + r + 8 * q;
      const float v = sacc[nt][r];
      if (j >= i) triu[base + (size_t)(i * NN - (i * (i - 1)) / 2 + (j - i))] = (__bf16)v;
      if (j <= i) tril[base + (size_t)((i * (i + 1)) / 2 + j)] = (__bf16)v;
    }
  }
}

// ---------------------------------------------------------------------------
// Kernel 2: one-shot f32 -> bf16 conversion of a view weight matrix (F x KP).
// Exact grid: NF*KP/8 chunks of 8 elements, 256 threads/block.
// ---------------------------------------------------------------------------
__global__ __launch_bounds__(256) void w_convert_kernel(
    const float* __restrict__ src, __bf16* __restrict__ dst)
{
  const size_t i = ((size_t)blockIdx.x * 256 + threadIdx.x) * 8;
  const float4 f0 = *(const float4*)(src + i);
  const float4 f1 = *(const float4*)(src + i + 4);
  union { __bf16 h[8]; uint4 u; } tmp;
  tmp.h[0] = (__bf16)f0.x; tmp.h[1] = (__bf16)f0.y;
  tmp.h[2] = (__bf16)f0.z; tmp.h[3] = (__bf16)f0.w;
  tmp.h[4] = (__bf16)f1.x; tmp.h[5] = (__bf16)f1.y;
  tmp.h[6] = (__bf16)f1.z; tmp.h[7] = (__bf16)f1.w;
  *(uint4*)(dst + i) = tmp.u;
}

// ---------------------------------------------------------------------------
// Kernel 3: out(4096x4096,f32) = A_bf16(4096x2080) @ B_bf16(4096x2080)^T
// 128x128 block tile, 8 waves of 64x32, double-buffered LDS staged with
// GLOBAL_LOAD_ASYNC_TO_LDS_B128 (ASYNCcnt), bf16 WMMA (K-step 32).
// ---------------------------------------------------------------------------
__global__ __launch_bounds__(256) void view_gemm_kernel(
    const __bf16* __restrict__ A, const __bf16* __restrict__ Bw,
    float* __restrict__ out)
{
  __shared__ __align__(16) __bf16 As[2][128 * 40];  // 32 k + 8 pad per row
  __shared__ __align__(16) __bf16 Bs[2][128 * 40];

  const int t    = threadIdx.x;
  const int wid  = t >> 5, lane = t & 31;
  const int rl   = lane & 15, q = lane >> 4;
  const int wm   = wid >> 2;          // 0..1 : 64-row strip
  const int wn   = wid & 3;           // 0..3 : 32-col strip
  const size_t bm = (size_t)blockIdx.x * 128;
  const size_t bn = (size_t)blockIdx.y * 128;

  const int r0 = t >> 2;   // staging row 0..63 (and +64)
  const int cc = t & 3;    // 16B chunk within the 32-k row

  // Per-thread staging addresses. Generic LDS pointers map addr[31:0] to the
  // LDS offset, so truncation gives the byte address the async op needs.
  const unsigned ldsA0 = (unsigned)(size_t)&As[0][r0 * 40 + cc * 8];
  const unsigned ldsA1 = (unsigned)(size_t)&As[1][r0 * 40 + cc * 8];
  const unsigned ldsB0 = (unsigned)(size_t)&Bs[0][r0 * 40 + cc * 8];
  const unsigned ldsB1 = (unsigned)(size_t)&Bs[1][r0 * 40 + cc * 8];
  const unsigned ldsRowStride = 64u * 40u * 2u;          // +64 rows, bytes
  const __bf16* gA = A  + (bm + r0) * (size_t)KP + cc * 8;
  const __bf16* gB = Bw + (bn + r0) * (size_t)KP + cc * 8;
  const size_t gRowStride = 64 * (size_t)KP;

  // prologue: stage K-step 0 into buffer 0 (4 async ops per wave)
  async_copy_b128(ldsA0,                gA);
  async_copy_b128(ldsA0 + ldsRowStride, gA + gRowStride);
  async_copy_b128(ldsB0,                gB);
  async_copy_b128(ldsB0 + ldsRowStride, gB + gRowStride);

  const v8f vz = {0.f, 0.f, 0.f, 0.f, 0.f, 0.f, 0.f, 0.f};
  v8f acc[4][2] = {{vz, vz}, {vz, vz}, {vz, vz}, {vz, vz}};

  for (int ks = 0; ks < KSTEPS; ++ks) {
    const int cur = ks & 1;
    // stage next K-step into the other buffer (consumed in the previous iteration;
    // the end-of-iteration barrier below makes it safe to overwrite)
    if (ks + 1 < KSTEPS) {
      const size_t ko = (size_t)(ks + 1) * 32;
      const unsigned la = cur ? ldsA0 : ldsA1;
      const unsigned lb = cur ? ldsB0 : ldsB1;
      async_copy_b128(la,                gA + ko);
      async_copy_b128(la + ldsRowStride, gA + gRowStride + ko);
      async_copy_b128(lb,                gB + ko);
      async_copy_b128(lb + ldsRowStride, gB + gRowStride + ko);
      wait_async_le4();   // retire this wave's 4 ops for the current buffer
    } else {
      wait_async_0();
    }
    __syncthreads();      // all waves' DMA chunks for cur buffer now visible

    const __bf16* Ab = cur ? &As[1][0] : &As[0][0];
    const __bf16* Bb = cur ? &Bs[1][0] : &Bs[0][0];

    v16bf af[4], bfr[2];
#pragma unroll
    for (int mt = 0; mt < 4; ++mt) {   // A frag: K chunks [8q,8q+8) and [16+8q,..)
      const int ra = wm * 64 + mt * 16 + rl;
      union { v16bf v; uint4 x[2]; } tmp;
      tmp.x[0] = *(const uint4*)(Ab + ra * 40 + q * 8);
      tmp.x[1] = *(const uint4*)(Ab + ra * 40 + 16 + q * 8);
      af[mt] = tmp.v;
    }
#pragma unroll
    for (int nt = 0; nt < 2; ++nt) {   // B frag: K chunks [8q,8q+8) and [16+8q,..)
      const int rb = wn * 32 + nt * 16 + rl;
      union { v16bf v; uint4 x[2]; } tmp;
      tmp.x[0] = *(const uint4*)(Bb + rb * 40 + q * 8);
      tmp.x[1] = *(const uint4*)(Bb + rb * 40 + 16 + q * 8);
      bfr[nt] = tmp.v;
    }
#pragma unroll
    for (int mt = 0; mt < 4; ++mt)
#pragma unroll
      for (int nt = 0; nt < 2; ++nt)
        acc[mt][nt] = __builtin_amdgcn_wmma_f32_16x16x32_bf16(
            false, af[mt], false, bfr[nt], (short)0, acc[mt][nt], false, false);

    __syncthreads();      // all waves done reading cur buffer -> restageable
  }

  // store C tiles (f32)
#pragma unroll
  for (int mt = 0; mt < 4; ++mt)
#pragma unroll
    for (int nt = 0; nt < 2; ++nt) {
      const size_t row = bm + (size_t)(wm * 64 + mt * 16);
      const size_t col = bn + (size_t)(wn * 32 + nt * 16 + rl);
#pragma unroll
      for (int r = 0; r < 8; ++r)
        out[(row + r + 8 * q) * (size_t)NF + col] = acc[mt][nt][r];
    }
}

// ---------------------------------------------------------------------------
extern "C" void kernel_launch(void* const* d_in, const int* in_sizes, int n_in,
                              void* d_out, int out_size, void* d_ws, size_t ws_size,
                              hipStream_t stream) {
  const float* feat  = (const float*)d_in[0];   // (B, N, D) f32
  const float* kpW   = (const float*)d_in[1];   // (D, D)   f32
  const float* Wtriu = (const float*)d_in[2];   // (F, KP)  f32
  const float* Wtril = (const float*)d_in[3];   // (F, KP)  f32

  float* out  = (float*)d_out;
  float* flat = out;                                   // (B, N*D)
  float* v1   = out + (size_t)NB * NN * ND;            // (B, F)
  float* v2   = v1 + (size_t)NB * NF;                  // (B, F)

  const size_t MAT = (size_t)NB * KP;                  // == NF * KP
  __bf16* triu  = (__bf16*)d_ws;                       // (B, KP) bf16
  __bf16* tril  = triu + MAT;                          // (B, KP) bf16
  __bf16* WtB16 = tril + MAT;                          // (F, KP) bf16
  __bf16* WlB16 = WtB16 + MAT;                         // (F, KP) bf16

  const int convBlocks = (int)(MAT / 8 / 256);         // exact: 4160
  w_convert_kernel<<<convBlocks, 256, 0, stream>>>(Wtriu, WtB16);
  w_convert_kernel<<<convBlocks, 256, 0, stream>>>(Wtril, WlB16);

  kp_extract_kernel<<<NB, 128, 0, stream>>>(feat, kpW, flat, triu, tril);

  dim3 g(32, 32, 1);                                   // 4096/128 each way
  view_gemm_kernel<<<g, 256, 0, stream>>>(triu, WtB16, v1);
  view_gemm_kernel<<<g, 256, 0, stream>>>(tril, WlB16, v2);
}
